// AttentionLayerBase_5145370821411
// MI455X (gfx1250) — matrix operand order repaired and checked by reading, not compiled, not run on hardware
//
#include <hip/hip_runtime.h>
#include <hip/hip_bf16.h>
#include <cstdint>

// ---------------------------------------------------------------------------
// CDNA5 (gfx1250, wave32) fused relative-position attention layer.
//   Qh/Kh/Vh/Rh/Oh intermediates kept in f16 in d_ws; all matmuls on
//   v_wmma_f32_16x16x32_f16; flash-style streaming softmax.
// ---------------------------------------------------------------------------

typedef _Float16 f16;
typedef __attribute__((ext_vector_type(16))) _Float16 v16h;
typedef __attribute__((ext_vector_type(8)))  float    v8f;

#define WMMA_F32_F16(a, b, c)                                                  \
  __builtin_amdgcn_wmma_f32_16x16x32_f16(false, (a), false, (b), (short)0,     \
                                         (c), false, false)

static constexpr int NBASIS = 10;

// A-fragment (16x32 f16) from LDS tile stored [row][k] with given f16 stride.
// Per ISA 7.12.2: lane l (g=l>>4, r=l&15) holds row r; element e maps to
// K = g*8 + e (e<8) or K = 16 + g*8 + (e-8).
__device__ __forceinline__ v16h load_a_frag(const f16* base, int row0, int k0,
                                            int stride) {
  const int lane = threadIdx.x & 31;
  const int g = lane >> 4, r = lane & 15;
  const f16* p = base + (size_t)(row0 + r) * stride + k0 + g * 8;
  v16h a;
#pragma unroll
  for (int e = 0; e < 8; ++e) a[e] = p[e];
#pragma unroll
  for (int e = 0; e < 8; ++e) a[8 + e] = p[16 + e];
  return a;
}

// B-fragment (32x16 f16) from LDS tile stored transposed: [col][k].
// Lane l (g=l>>4, c=l&15) holds column c; element e maps to
// K = g*8 + e (e<8) or K = 16 + g*8 + (e-8).
__device__ __forceinline__ v16h load_b_frag(const f16* base, int col0, int k0,
                                            int stride) {
  const int lane = threadIdx.x & 31;
  const int g = lane >> 4, c = lane & 15;
  const f16* p = base + (size_t)(col0 + c) * stride + k0 + g * 8;
  v16h b;
#pragma unroll
  for (int e = 0; e < 8; ++e) b[e] = p[e];
#pragma unroll
  for (int e = 0; e < 8; ++e) b[8 + e] = p[16 + e];
  return b;
}

// ---------------------------------------------------------------------------
// Generic GEMM: Out[M,N] = A[M,Kd] * B[Kd,N] * scale + bias
//   A fp32 or f16 (row-major); B fp32 row-major; Out f16 or fp32.
//   128x128 block tile, 8 waves in 4(M) x 2(N), each wave = 2x4 WMMA tiles.
// ---------------------------------------------------------------------------
template <bool A_HALF, bool OUT_HALF>
__global__ __launch_bounds__(256) void gemm_wmma_kernel(
    const void* __restrict__ Av, const float* __restrict__ Bm,
    const float* __restrict__ bias, float scale, void* __restrict__ Outv,
    int M, int N, int Kd) {
  (void)M;
  __shared__ f16 Ah[128][36];  // [row][k]   (k-chunk of 32, pad to 36)
  __shared__ f16 Bt[128][36];  // [col][k]   (transposed during staging)

  const int tid = threadIdx.x;
  const int n0 = blockIdx.x * 128;
  const int m0 = blockIdx.y * 128;
  const int w = tid >> 5, lane = tid & 31, g = lane >> 4, c = lane & 15;
  const int wm = w & 3, wn = w >> 2;  // wave tile: rows wm*32, cols wn*64

  v8f zf = {};
  v8f acc[2][4];
#pragma unroll
  for (int a = 0; a < 2; ++a)
#pragma unroll
    for (int b = 0; b < 4; ++b) acc[a][b] = zf;

  for (int k0 = 0; k0 < Kd; k0 += 32) {
    __syncthreads();  // previous iteration's fragment reads complete
    {                 // stage A tile: 128 rows x 32 k, 2 threads/row
      const int r = tid >> 1, kk = (tid & 1) * 16;
      if constexpr (A_HALF) {
        const f16* src = (const f16*)Av + (size_t)(m0 + r) * Kd + k0 + kk;
#pragma unroll
        for (int i = 0; i < 16; ++i) Ah[r][kk + i] = src[i];
      } else {
        const float* src = (const float*)Av + (size_t)(m0 + r) * Kd + k0 + kk;
#pragma unroll
        for (int i = 0; i < 16; ++i) Ah[r][kk + i] = (f16)src[i];
      }
    }
    {  // stage B tile transposed: read [k][n] coalesced, write [n][k]
      const int n4 = (tid & 31) * 4;
      const int kr = tid >> 5;
#pragma unroll
      for (int kp = 0; kp < 4; ++kp) {
        const int k = kr + kp * 8;
        const float* src = Bm + (size_t)(k0 + k) * N + n0 + n4;
#pragma unroll
        for (int i = 0; i < 4; ++i) {
          const float v = (n0 + n4 + i < N) ? src[i] : 0.f;
          Bt[n4 + i][k] = (f16)v;
        }
      }
    }
    __syncthreads();

    const v16h a0 = load_a_frag(&Ah[0][0], wm * 32, 0, 36);
    const v16h a1 = load_a_frag(&Ah[0][0], wm * 32 + 16, 0, 36);
#pragma unroll
    for (int kn = 0; kn < 4; ++kn) {
      const v16h b = load_b_frag(&Bt[0][0], wn * 64 + kn * 16, 0, 36);
      acc[0][kn] = WMMA_F32_F16(a0, b, acc[0][kn]);
      acc[1][kn] = WMMA_F32_F16(a1, b, acc[1][kn]);
    }
  }

  // Epilogue: C layout — lane (g,c): VGPR v -> row v+8g, col c.
  f16* hout = (f16*)Outv;
  float* fout = (float*)Outv;
#pragma unroll
  for (int kn = 0; kn < 4; ++kn) {
    const int col = n0 + wn * 64 + kn * 16 + c;
    const bool inN = col < N;
    const float bv = (bias != nullptr && inN) ? bias[col] : 0.f;
#pragma unroll
    for (int km = 0; km < 2; ++km)
#pragma unroll
      for (int v = 0; v < 8; ++v) {
        const int row = m0 + wm * 32 + km * 16 + v + 8 * g;
        const float val = acc[km][kn][v] * scale + bv;
        if (inN) {
          if constexpr (OUT_HALF)
            hout[(size_t)row * N + col] = (f16)val;
          else
            fout[(size_t)row * N + col] = val;
        }
      }
  }
}

// ---------------------------------------------------------------------------
// Flash attention with banded relative-position bias.
//   Grid: (T/128 query tiles, 16 heads); 8 waves, each owns 16 query rows.
//   S = Q*K^T (Q pre-scaled by 1/sqrt(d)), +extra(i,j)=sum_n R[i,n]*bnd[n,i-j],
//   causal mask, online softmax, O += P*V.
// ---------------------------------------------------------------------------
__global__ __launch_bounds__(256) void attn_wmma_kernel(
    const f16* __restrict__ Qh, const f16* __restrict__ Kh,
    const f16* __restrict__ Vh, const f16* __restrict__ Rh,
    const float* __restrict__ bnd, f16* __restrict__ Oh,
    int T, int E, int RW, int ML) {
  __shared__ f16 Qs[128][72];       // [qrow][d]
  __shared__ f16 Ks[64][72];        // [key][d]     (B-layout for S=Q*K^T)
  __shared__ f16 Vt[64][72];        // [d][key]     (B-layout for O=P*V)
  __shared__ float Rs[128][12];     // per-head basis coefficients
  __shared__ f16 Ps[8][16][36];     // per-wave P relayout scratch (16x32)

  const int tid = threadIdx.x;
  const int h = blockIdx.y;
  const int q0 = blockIdx.x * 128;

  {  // stage Q tile (already scaled by 1/sqrt(d))
    const int r = tid >> 1, doff = (tid & 1) * 32;
    const f16* src = Qh + (size_t)(q0 + r) * E + h * 64 + doff;
#pragma unroll
    for (int i = 0; i < 32; ++i) Qs[r][doff + i] = src[i];
  }
  for (int idx = tid; idx < 128 * NBASIS; idx += 256) {
    const int r = idx / NBASIS, n = idx % NBASIS;
    Rs[r][n] = (float)Rh[(size_t)(q0 + r) * RW + h * NBASIS + n];
  }
  __syncthreads();

  const int w = tid >> 5, lane = tid & 31, g = lane >> 4, c = lane & 15;
  const int w16 = w * 16, q0w = q0 + w16;

  const v16h qa0 = load_a_frag(&Qs[0][0], w16, 0, 72);
  const v16h qa1 = load_a_frag(&Qs[0][0], w16, 32, 72);

  v8f zf = {};
  v8f Ofr[4];
#pragma unroll
  for (int dt = 0; dt < 4; ++dt) Ofr[dt] = zf;
  float mrow[8], lrow[8];
#pragma unroll
  for (int v = 0; v < 8; ++v) { mrow[v] = -3.0e38f; lrow[v] = 0.f; }

  for (int kb = 0; kb <= q0 + 127; kb += 64) {  // causal key-chunk loop
    __syncthreads();
    {  // stage K [key][d] and V transposed [d][key]
      const int r = tid >> 2, doff = (tid & 3) * 16;
      const f16* ksrc = Kh + (size_t)(kb + r) * E + h * 64 + doff;
      const f16* vsrc = Vh + (size_t)(kb + r) * E + h * 64 + doff;
#pragma unroll
      for (int i = 0; i < 16; ++i) Ks[r][doff + i] = ksrc[i];
#pragma unroll
      for (int i = 0; i < 16; ++i) Vt[doff + i][r] = vsrc[i];
    }
    __syncthreads();

#pragma unroll
    for (int half = 0; half < 2; ++half) {  // 32 keys per softmax step
      const int jb = kb + half * 32;
      if (jb > q0w + 15) continue;  // wave-uniform: fully masked

      v8f s[2];
      s[0] = zf; s[1] = zf;
#pragma unroll
      for (int t = 0; t < 2; ++t) {
        const v16h bk0 = load_b_frag(&Ks[0][0], half * 32 + t * 16, 0, 72);
        s[t] = WMMA_F32_F16(qa0, bk0, s[t]);
        const v16h bk1 = load_b_frag(&Ks[0][0], half * 32 + t * 16, 32, 72);
        s[t] = WMMA_F32_F16(qa1, bk1, s[t]);
      }

      // relative-position bias + causal mask, per C-fragment element
      float sb[2][8], mt[8];
#pragma unroll
      for (int v = 0; v < 8; ++v) mt[v] = -3.0e38f;
#pragma unroll
      for (int t = 0; t < 2; ++t)
#pragma unroll
        for (int v = 0; v < 8; ++v) {
          const int rloc = w16 + v + 8 * g;  // row within 128-tile
          const int i = q0 + rloc;
          const int j = jb + t * 16 + c;
          float sv;
          if (j > i) {
            sv = -1.0e30f;
          } else {
            const int rel = i - j;
            float ex = 0.f;
            if (rel < ML) {
#pragma unroll
              for (int n = 0; n < NBASIS; ++n)
                ex = fmaf(Rs[rloc][n], bnd[n * ML + rel], ex);
            }
            sv = s[t][v] + ex;
          }
          sb[t][v] = sv;
          mt[v] = fmaxf(mt[v], sv);
        }

      // row max across the 16 lanes of each half-wave group
#pragma unroll
      for (int v = 0; v < 8; ++v) {
#pragma unroll
        for (int off = 8; off >= 1; off >>= 1)
          mt[v] = fmaxf(mt[v], __shfl_xor(mt[v], off, 32));
      }
      float alpha[8], rsum[8];
#pragma unroll
      for (int v = 0; v < 8; ++v) {
        const float mn = fmaxf(mrow[v], mt[v]);
        alpha[v] = __expf(mrow[v] - mn);
        mrow[v] = mn;
        rsum[v] = 0.f;
      }
#pragma unroll
      for (int t = 0; t < 2; ++t)
#pragma unroll
        for (int v = 0; v < 8; ++v) {
          const float p = __expf(sb[t][v] - mrow[v]);
          rsum[v] += p;
          Ps[w][v + 8 * g][t * 16 + c] = (f16)p;  // C-layout -> LDS
        }
#pragma unroll
      for (int v = 0; v < 8; ++v) {
#pragma unroll
        for (int off = 8; off >= 1; off >>= 1)
          rsum[v] += __shfl_xor(rsum[v], off, 32);
        lrow[v] = lrow[v] * alpha[v] + rsum[v];
      }
#pragma unroll
      for (int dt = 0; dt < 4; ++dt)
#pragma unroll
        for (int v = 0; v < 8; ++v) Ofr[dt][v] *= alpha[v];

      // cross-lane LDS RAW inside the wave: drain DS before A-frag reload
      asm volatile("s_wait_dscnt 0x0" ::: "memory");
      const v16h pa = load_a_frag(&Ps[w][0][0], 0, 0, 36);
#pragma unroll
      for (int dt = 0; dt < 4; ++dt) {
        const v16h bv = load_b_frag(&Vt[0][0], dt * 16, half * 32, 72);
        Ofr[dt] = WMMA_F32_F16(pa, bv, Ofr[dt]);
      }
    }
  }

  // normalize and write head output slice (f16)
#pragma unroll
  for (int dt = 0; dt < 4; ++dt)
#pragma unroll
    for (int v = 0; v < 8; ++v) {
      const int i = q0 + w16 + v + 8 * g;
      const float o = Ofr[dt][v] / lrow[v];
      Oh[(size_t)i * E + h * 64 + dt * 16 + c] = (f16)o;
    }
}

// ---------------------------------------------------------------------------
extern "C" void kernel_launch(void* const* d_in, const int* in_sizes, int n_in,
                              void* d_out, int out_size, void* d_ws,
                              size_t ws_size, hipStream_t stream) {
  (void)n_in; (void)out_size; (void)ws_size;
  const float* X      = (const float*)d_in[0];
  const float* W_q    = (const float*)d_in[1];
  const float* b_q    = (const float*)d_in[2];
  const float* W_k    = (const float*)d_in[3];
  const float* W_v    = (const float*)d_in[4];
  const float* W_proj = (const float*)d_in[5];
  const float* b_proj = (const float*)d_in[6];
  const float* W_r    = (const float*)d_in[7];
  const float* b_r    = (const float*)d_in[8];
  const float* b_nd   = (const float*)d_in[9];

  const int E = 1024, QK = 1024, H = 16;
  const int RW = NBASIS * H;            // 160
  const int T  = in_sizes[0] / E;       // 2048
  const int ML = in_sizes[9] / NBASIS;  // 2048

  char* ws = (char*)d_ws;
  f16* Qh = (f16*)ws; ws += (size_t)T * QK * sizeof(f16);
  f16* Kh = (f16*)ws; ws += (size_t)T * QK * sizeof(f16);
  f16* Vh = (f16*)ws; ws += (size_t)T * QK * sizeof(f16);
  f16* Rh = (f16*)ws; ws += (size_t)T * RW * sizeof(f16);
  f16* Oh = (f16*)ws;

  const dim3 blk(256);
  const float qscale = 0.125f;  // 1/sqrt(64), folded into Q projection
  gemm_wmma_kernel<false, true><<<dim3(QK / 128, T / 128), blk, 0, stream>>>(
      X, W_q, b_q, qscale, Qh, T, QK, E);
  gemm_wmma_kernel<false, true><<<dim3(QK / 128, T / 128), blk, 0, stream>>>(
      X, W_k, nullptr, 1.f, Kh, T, QK, E);
  gemm_wmma_kernel<false, true><<<dim3(QK / 128, T / 128), blk, 0, stream>>>(
      X, W_v, nullptr, 1.f, Vh, T, QK, E);
  gemm_wmma_kernel<false, true><<<dim3((RW + 127) / 128, T / 128), blk, 0,
                                  stream>>>(X, W_r, b_r, 1.f, Rh, T, RW, E);
  attn_wmma_kernel<<<dim3(T / 128, H), blk, 0, stream>>>(Qh, Kh, Vh, Rh, b_nd,
                                                         Oh, T, E, RW, ML);
  gemm_wmma_kernel<true, false><<<dim3(E / 128, T / 128), blk, 0, stream>>>(
      Oh, W_proj, b_proj, 1.f, d_out, T, E, QK);
}
